// MultiLayerLSTM_13159779795582
// MI455X (gfx1250) — matrix operand order repaired and checked by reading, not compiled, not run on hardware
//
#include <hip/hip_runtime.h>
#include <hip/hip_bf16.h>

// ---------------------------------------------------------------------------
// MultiLayerLSTM for MI455X (gfx1250, wave32, WMMA + async LDS copies)
//   B=64, T=512, D=512, H=1024, 2 layers.
//   gates = x_t @ W + h_{t-1} @ U + b ; fused activations + cell update.
// ---------------------------------------------------------------------------

#define LSTM_B   64
#define LSTM_T   512
#define LSTM_D   512
#define LSTM_H   1024
#define N4H      4096   // 4*H

typedef __attribute__((ext_vector_type(16))) __bf16 bf16x16;
typedef __attribute__((ext_vector_type(8)))  __bf16 bf16x8;
typedef __attribute__((ext_vector_type(8)))  float  f32x8;

struct U4x2 { uint4 lo, hi; };

static __device__ inline bf16x16 pack_bf16x16(uint4 lo, uint4 hi) {
    U4x2 t; t.lo = lo; t.hi = hi;
    return __builtin_bit_cast(bf16x16, t);
}

// gfx1250 async global->LDS copy (ASYNCcnt tracked, no VGPR round trip).
static __device__ inline void async_load_b128(unsigned lds_addr, const void* gptr) {
    asm volatile("global_load_async_to_lds_b128 %0, %1, off"
                 :: "v"(lds_addr), "v"((unsigned long long)(uintptr_t)gptr)
                 : "memory");
}

static __device__ inline void wait_async0() {
#if __has_builtin(__builtin_amdgcn_s_wait_asynccnt)
    __builtin_amdgcn_s_wait_asynccnt(0);
#else
    asm volatile("s_wait_asynccnt 0" ::: "memory");
#endif
}

// ---------------------------------------------------------------------------
// Weight conversion: src fp32 (K, 4H) row-major  ->  dst bf16 (4H', K)
// with gate interleave: dst row r = 4*j + g corresponds to src column g*H + j.
// ---------------------------------------------------------------------------
__global__ __launch_bounds__(256) void convert_weight_kernel(
    const float* __restrict__ src, __bf16* __restrict__ dst, int K)
{
    __shared__ float tile[32][33];
    const int c0  = blockIdx.x * 32;
    const int k0  = blockIdx.y * 32;
    const int tid = threadIdx.x;

    #pragma unroll
    for (int q = 0; q < 4; ++q) {
        int idx = tid + q * 256;
        int cl = idx & 31, kl = idx >> 5;
        tile[cl][kl] = src[(long)(k0 + kl) * N4H + c0 + cl];
    }
    __syncthreads();
    #pragma unroll
    for (int q = 0; q < 4; ++q) {
        int idx = tid + q * 256;
        int kl = idx & 31, cl = idx >> 5;
        int c = c0 + cl;
        int j = c & (LSTM_H - 1);
        int g = c >> 10;
        int r = j * 4 + g;
        dst[(long)r * K + k0 + kl] = (__bf16)tile[cl][kl];
    }
}

// ---------------------------------------------------------------------------
// Fused LSTM step, templated on the two K depths (compile-time trip counts).
//   A0/A1: fp32 (64, K) row stride lda ; B0/B1: bf16 (4H gate-interleaved, K)
// Block: 64(M) x 64(N) tile, 8 waves; K-chunk = 64 (4 WMMA / wave / chunk).
// B tiles staged with global_load_async_to_lds_b128 into double-buffered LDS.
// ---------------------------------------------------------------------------
template<int K0, int K1>
__global__ __launch_bounds__(256) void lstm_step_kernel(
    const float* __restrict__ A0, long lda0, const __bf16* __restrict__ B0,
    const float* __restrict__ A1, long lda1, const __bf16* __restrict__ B1,
    const float* __restrict__ bias,
    const float* __restrict__ c_in, float* __restrict__ c_out,
    float* __restrict__ h_out, float* __restrict__ hseq, long ld_hseq)
{
    constexpr int NC0 = K0 / 64;
    constexpr int NC1 = K1 / 64;
    constexpr int NC  = NC0 + NC1;

    __shared__ __align__(16) __bf16 As[2][64][72];   // 64 x 64K, pitch 72
    __shared__ __align__(16) __bf16 Bs[2][64][72];
    __shared__ float Gs[64][68];

    const int tid  = threadIdx.x;
    const int lane = tid & 31;
    const int wave = tid >> 5;
    const int mt   = (wave & 3) * 16;   // wave M base
    const int nh   = (wave >> 2) * 32;  // wave N base (two 16-col tiles)
    const int r0   = blockIdx.x * 64;   // gate-interleaved N row base

    // per-lane WMMA operand addressing (contiguous k = 16*(lane/16) + e, same order for A and B)
    const int arow = lane & 15;
    const int akb  = (lane >> 4) * 16;  // A K-halves {0..15}/{16..31}
    const int bko  = (lane >> 4) * 16;  // B K-halves {0..15}/{16..31}

    f32x8 acc0 = {};
    f32x8 acc1 = {};

    // ---- staging: A = fp32->bf16 via VGPRs, B = async direct to LDS ----
    auto stage = [&](int i, int buf) {
        const float*  A;  long lda;  const __bf16* Bt;  int K, kk;
        if (i < NC0) { A = A0; lda = lda0; Bt = B0; K = K0; kk = i * 64; }
        else         { A = A1; lda = lda1; Bt = B1; K = K1; kk = (i - NC0) * 64; }

        // A: 16 contiguous floats / thread -> 2x ds_store_b128 of bf16
        {
            int row = tid >> 2, seg = (tid & 3) * 16;
            const float4* src = (const float4*)(A + (long)row * lda + kk + seg);
            float4 f0 = src[0], f1 = src[1], f2 = src[2], f3 = src[3];
            bf16x8 va, vb;
            va[0] = (__bf16)f0.x; va[1] = (__bf16)f0.y;
            va[2] = (__bf16)f0.z; va[3] = (__bf16)f0.w;
            va[4] = (__bf16)f1.x; va[5] = (__bf16)f1.y;
            va[6] = (__bf16)f1.z; va[7] = (__bf16)f1.w;
            vb[0] = (__bf16)f2.x; vb[1] = (__bf16)f2.y;
            vb[2] = (__bf16)f2.z; vb[3] = (__bf16)f2.w;
            vb[4] = (__bf16)f3.x; vb[5] = (__bf16)f3.y;
            vb[6] = (__bf16)f3.z; vb[7] = (__bf16)f3.w;
            *(bf16x8*)(&As[buf][row][seg])     = va;
            *(bf16x8*)(&As[buf][row][seg + 8]) = vb;
        }
        // B: 64 rows x 64 bf16 = 512 x 16B blocks; 2 async b128 per thread
        #pragma unroll
        for (int u = 0; u < 2; ++u) {
            int idx = tid + u * 256;
            int row = idx >> 3;
            int blk = (idx & 7) * 8;
            async_load_b128((unsigned)(uintptr_t)(&Bs[buf][row][blk]),
                            Bt + (long)(r0 + row) * K + kk + blk);
        }
    };

    stage(0, 0);
    wait_async0();
    __syncthreads();

    for (int i = 0; i < NC; ++i) {
        const int buf = i & 1;
        if (i + 1 < NC) stage(i + 1, buf ^ 1);   // overlap with compute below

        #pragma unroll
        for (int s = 0; s < 64; s += 32) {
            uint4 alo = *(const uint4*)(&As[buf][mt + arow][s + akb]);
            uint4 ahi = *(const uint4*)(&As[buf][mt + arow][s + 8 + akb]);
            bf16x16 a = pack_bf16x16(alo, ahi);

            uint4 b0lo = *(const uint4*)(&Bs[buf][nh + arow][s + bko]);
            uint4 b0hi = *(const uint4*)(&Bs[buf][nh + arow][s + bko + 8]);
            uint4 b1lo = *(const uint4*)(&Bs[buf][nh + 16 + arow][s + bko]);
            uint4 b1hi = *(const uint4*)(&Bs[buf][nh + 16 + arow][s + bko + 8]);

            acc0 = __builtin_amdgcn_wmma_f32_16x16x32_bf16(
                       false, a, false, pack_bf16x16(b0lo, b0hi),
                       (short)0, acc0, false, false);
            acc1 = __builtin_amdgcn_wmma_f32_16x16x32_bf16(
                       false, a, false, pack_bf16x16(b1lo, b1hi),
                       (short)0, acc1, false, false);
        }

        wait_async0();     // next buffer's async B tiles landed
        __syncthreads();   // and everyone is done reading the current buffers
    }

    // ---- spill accumulators (C layout: lanes 0-15 M=v, lanes 16-31 M=v+8) ----
    {
        int crow = (lane >> 4) * 8;
        int ccol = lane & 15;
        #pragma unroll
        for (int v = 0; v < 8; ++v) {
            Gs[mt + crow + v][nh + ccol]      = acc0[v];
            Gs[mt + crow + v][nh + 16 + ccol] = acc1[v];
        }
    }
    __syncthreads();

    // ---- fused activations + cell/hidden update (64 batch x 16 units) ----
    const int j0 = blockIdx.x * 16;
    #pragma unroll
    for (int q = 0; q < 4; ++q) {
        int p  = tid + q * 256;
        int m  = p >> 4;
        int jl = p & 15;
        int j  = j0 + jl;

        float gi = Gs[m][jl * 4 + 0] + bias[j];
        float gf = Gs[m][jl * 4 + 1] + bias[LSTM_H + j];
        float gg = Gs[m][jl * 4 + 2] + bias[2 * LSTM_H + j];
        float go = Gs[m][jl * 4 + 3] + bias[3 * LSTM_H + j];

        gi = 1.0f / (1.0f + __expf(-gi));
        gf = 1.0f / (1.0f + __expf(-gf));
        gg = tanhf(gg);
        go = 1.0f / (1.0f + __expf(-go));

        float cp = c_in[(long)m * LSTM_H + j];
        float cn = gf * cp + gi * gg;
        float h  = go * tanhf(cn);

        c_out[(long)m * LSTM_H + j] = cn;
        h_out[(long)m * LSTM_H + j] = h;
        if (hseq) hseq[(long)m * ld_hseq + j] = h;
    }
}

// ---------------------------------------------------------------------------
// Host-side orchestration (graph-capture safe).
// ---------------------------------------------------------------------------
extern "C" void kernel_launch(void* const* d_in, const int* in_sizes, int n_in,
                              void* d_out, int out_size, void* d_ws, size_t ws_size,
                              hipStream_t stream)
{
    (void)in_sizes; (void)n_in; (void)out_size; (void)ws_size;

    const float* input_seq = (const float*)d_in[0]; // (64, 512, 512)
    const float* h_0       = (const float*)d_in[1]; // (2, 64, 1024)
    const float* c_0       = (const float*)d_in[2]; // (2, 64, 1024)
    const float* W0        = (const float*)d_in[3]; // (512, 4096)
    const float* U0        = (const float*)d_in[4]; // (1024, 4096)
    const float* b0        = (const float*)d_in[5]; // (4096,)
    const float* W1        = (const float*)d_in[6]; // (1024, 4096)
    const float* U1        = (const float*)d_in[7]; // (1024, 4096)
    const float* b1        = (const float*)d_in[8]; // (4096,)

    // ---- workspace layout ----
    __bf16* W0t = (__bf16*)d_ws;                       // 4096 x 512
    __bf16* U0t = W0t + (long)N4H * LSTM_D;            // 4096 x 1024
    __bf16* W1t = U0t + (long)N4H * LSTM_H;            // 4096 x 1024
    __bf16* U1t = W1t + (long)N4H * LSTM_H;            // 4096 x 1024
    float*  st  = (float*)(U1t + (long)N4H * LSTM_H);
    const long SB = (long)LSTM_B * LSTM_H;             // 65536 floats
    float* h0b = st;                                   // [2][B*H] ping-pong
    float* c0b = h0b + 2 * SB;
    float* h1b = c0b + 2 * SB;
    float* c1b = h1b + 2 * SB;

    float* xout = (float*)d_out;                                 // (64,512,1024)
    float* hfin = xout + (long)LSTM_B * LSTM_T * LSTM_H;
    float* cfin = hfin + SB;

    // ---- 1) weight convert + transpose + gate-interleave (bf16) ----
    convert_weight_kernel<<<dim3(N4H / 32, LSTM_D / 32), 256, 0, stream>>>(W0, W0t, LSTM_D);
    convert_weight_kernel<<<dim3(N4H / 32, LSTM_H / 32), 256, 0, stream>>>(U0, U0t, LSTM_H);
    convert_weight_kernel<<<dim3(N4H / 32, LSTM_H / 32), 256, 0, stream>>>(W1, W1t, LSTM_H);
    convert_weight_kernel<<<dim3(N4H / 32, LSTM_H / 32), 256, 0, stream>>>(U1, U1t, LSTM_H);

    // ---- 2) initialize recurrent state (buffer index 0) ----
    hipMemcpyAsync(h0b, h_0,      SB * sizeof(float), hipMemcpyDeviceToDevice, stream);
    hipMemcpyAsync(h1b, h_0 + SB, SB * sizeof(float), hipMemcpyDeviceToDevice, stream);
    hipMemcpyAsync(c0b, c_0,      SB * sizeof(float), hipMemcpyDeviceToDevice, stream);
    hipMemcpyAsync(c1b, c_0 + SB, SB * sizeof(float), hipMemcpyDeviceToDevice, stream);

    // ---- 3) 512 timesteps x 2 layers ----
    const long ldx  = (long)LSTM_T * LSTM_D;   // input_seq batch stride
    const long ldh  = LSTM_H;
    const long ldhs = (long)LSTM_T * LSTM_H;   // output seq batch stride

    for (int t = 0; t < LSTM_T; ++t) {
        const int cur = t & 1, nxt = cur ^ 1;

        lstm_step_kernel<LSTM_D, LSTM_H><<<dim3(LSTM_H / 16), 256, 0, stream>>>(
            input_seq + (long)t * LSTM_D, ldx, W0t,
            h0b + cur * SB, ldh, U0t,
            b0, c0b + cur * SB, c0b + nxt * SB, h0b + nxt * SB,
            nullptr, 0);

        lstm_step_kernel<LSTM_H, LSTM_H><<<dim3(LSTM_H / 16), 256, 0, stream>>>(
            h0b + nxt * SB, ldh, W1t,
            h1b + cur * SB, ldh, U1t,
            b1, c1b + cur * SB, c1b + nxt * SB, h1b + nxt * SB,
            xout + (long)t * LSTM_H, ldhs);
    }

    // ---- 4) final h, c of layer 1 (state sits in buffer index 0 after 512) ----
    hipMemcpyAsync(hfin, h1b, SB * sizeof(float), hipMemcpyDeviceToDevice, stream);
    hipMemcpyAsync(cfin, c1b, SB * sizeof(float), hipMemcpyDeviceToDevice, stream);
}
